// SyncQwen3VLMoeSparseMoeBlock_28003186770595
// MI455X (gfx1250) — matrix operand order repaired and checked by reading, not compiled, not run on hardware
//
#include <hip/hip_runtime.h>
#include <hip/hip_bf16.h>
#include <stdint.h>

// ---------------------------------------------------------------------------
// MoE (8 experts, top-2, H=1024, I=1408, T=2048) for gfx1250.
// bf16 WMMA (v_wmma_f32_16x16x32_bf16) with fp32 accumulate; fp32->bf16
// conversion happens while staging tiles into LDS so HBM sees exactly one
// fp32 pass over the 138 MB of weights (~6.7 us floor at 23.3 TB/s).
// Down-projection A-tiles (already bf16 in workspace) are staged with the
// CDNA5 async global->LDS path (ASYNCcnt) into a double-buffered LDS tile.
// Fragment ds_loads are software-pipelined ahead of the WMMA that consumes
// the previous fragment so per-wave LDS latency overlaps the matrix pipe.
// ---------------------------------------------------------------------------

typedef __attribute__((ext_vector_type(16))) __bf16 v16bf;
typedef __attribute__((ext_vector_type(8)))  float  v8f;
typedef __attribute__((ext_vector_type(4)))  unsigned int u32x4;
typedef int v4i_vs __attribute__((vector_size(16)));   // matches builtin param

#define NUM_EXPERTS 8
#define TOPK 2
#define HIDDEN 1024
#define INTER 1408
#define T_TOK 2048
#define NPAIR (T_TOK * TOPK)

#define BM 64
#define BN 128
#define BK 32
#define LDT 40                 // LDS row pitch in bf16 elems (80B: 16B aligned, conflict-free)
#define MAX_MT (T_TOK / BM)    // 32 worst-case M-tiles per expert

// ----- CDNA5 async global->LDS (guarded; sync fallback keeps compile safe) --

#if defined(__has_builtin)
#if __has_builtin(__builtin_amdgcn_global_load_async_to_lds_b128)
#define HAVE_ASYNC_LDS 1
#endif
#endif

__device__ __forceinline__ void async_copy_b128(const unsigned short* g,
                                                unsigned short* l) {
#if defined(HAVE_ASYNC_LDS)
  __builtin_amdgcn_global_load_async_to_lds_b128(
      (__attribute__((address_space(1))) v4i_vs*)g,
      (__attribute__((address_space(3))) v4i_vs*)l, 0, 0);
#else
  *(u32x4*)l = *(const u32x4*)g;
#endif
}

#if defined(HAVE_ASYNC_LDS)
#if __has_builtin(__builtin_amdgcn_s_wait_asynccnt)
#define WAIT_ASYNC(n) __builtin_amdgcn_s_wait_asynccnt(n)
#else
#define WAIT_ASYNC(n) asm volatile("s_wait_asynccnt %0" ::"n"(n))
#endif
#else
#define WAIT_ASYNC(n)
#endif

// ----- helpers --------------------------------------------------------------

__device__ __forceinline__ unsigned short f2bf(float f) {
  unsigned int u = __builtin_bit_cast(unsigned int, f);
  unsigned int r = u + 0x7fffu + ((u >> 16) & 1u);   // round-to-nearest-even
  return (unsigned short)(r >> 16);
}

__device__ __forceinline__ unsigned int pack2(float a, float b) {
  return (unsigned int)f2bf(a) | ((unsigned int)f2bf(b) << 16);
}

struct FragBits { u32x4 x, y; };

// A-matrix 16x32 bf16 fragment per ISA layout:
// lane half 0: K 0..7 then 16..23 ; lane half 1: K 8..15 then 24..31.
__device__ __forceinline__ v16bf load_fragA(const unsigned short* p0, int lane) {
  int row = lane & 15, half = lane >> 4;
  const unsigned short* p = p0 + row * LDT + half * 8;
  FragBits t;
  t.x = *(const u32x4*)(p);        // K base..base+7
  t.y = *(const u32x4*)(p + 16);   // K base+16..base+23
  return __builtin_bit_cast(v16bf, t);
}

// B-matrix 32x16 bf16 fragment (col = lane%16): lane half 0: K 0..7 then 16..23;
// lane half 1: K 8..15 then 24..31. Tile stored transposed in LDS: Bs[col][k].
__device__ __forceinline__ v16bf load_fragB(const unsigned short* p0, int lane) {
  int col = lane & 15, half = lane >> 4;
  const unsigned short* p = p0 + col * LDT + half * 8;
  FragBits t;
  t.x = *(const u32x4*)(p);
  t.y = *(const u32x4*)(p + 16);
  return __builtin_bit_cast(v16bf, t);
}

__device__ __forceinline__ v8f wmma_bf16(v16bf a, v16bf b, v8f c) {
  return __builtin_amdgcn_wmma_f32_16x16x32_bf16(
      /*neg_a=*/false, a, /*neg_b=*/false, b,
      /*c_mod=*/(short)0, c, /*reuse_a=*/false, /*reuse_b=*/false);
}

// ----- routing kernels ------------------------------------------------------

__global__ void moe_init(int* cnt, int* cnt2) {
  int i = threadIdx.x;
  if (i < NUM_EXPERTS) { cnt[i] = 0; cnt2[i] = 0; }
}

__global__ __launch_bounds__(256) void moe_router(
    const float* __restrict__ x, const float* __restrict__ gate_w,
    int* __restrict__ cnt, int* __restrict__ sel, float* __restrict__ wsel) {
  __shared__ float gw[HIDDEN * NUM_EXPERTS];
  for (int i = threadIdx.x; i < HIDDEN * NUM_EXPERTS; i += 256) gw[i] = gate_w[i];
  __syncthreads();
  int t = blockIdx.x * 256 + threadIdx.x;
  if (t >= T_TOK) return;
  const float4* xr4 = (const float4*)(x + (size_t)t * HIDDEN);
  float acc[NUM_EXPERTS];
#pragma unroll
  for (int e = 0; e < NUM_EXPERTS; ++e) acc[e] = 0.f;
  for (int h4 = 0; h4 < HIDDEN / 4; ++h4) {
    float4 xv = xr4[h4];
    const float* g0 = &gw[(h4 * 4 + 0) * NUM_EXPERTS];
#pragma unroll
    for (int e = 0; e < NUM_EXPERTS; ++e)
      acc[e] += xv.x * g0[e] + xv.y * g0[NUM_EXPERTS + e] +
                xv.z * g0[2 * NUM_EXPERTS + e] + xv.w * g0[3 * NUM_EXPERTS + e];
  }
  float m = acc[0];
#pragma unroll
  for (int e = 1; e < NUM_EXPERTS; ++e) m = fmaxf(m, acc[e]);
  float p[NUM_EXPERTS];
#pragma unroll
  for (int e = 0; e < NUM_EXPERTS; ++e) p[e] = __expf(acc[e] - m);
  int i0 = 0;
#pragma unroll
  for (int e = 1; e < NUM_EXPERTS; ++e) if (p[e] > p[i0]) i0 = e;  // first max wins
  int i1 = (i0 == 0) ? 1 : 0;
#pragma unroll
  for (int e = 0; e < NUM_EXPERTS; ++e)
    if (e != i0 && p[e] > p[i1]) i1 = e;
  float s = p[i0] + p[i1];                       // softmax denom cancels in renorm
  sel[2 * t] = i0;  sel[2 * t + 1] = i1;
  wsel[2 * t] = p[i0] / s;  wsel[2 * t + 1] = p[i1] / s;
  atomicAdd(&cnt[i0], 1);
  atomicAdd(&cnt[i1], 1);
}

__global__ void moe_scan(const int* __restrict__ cnt, int* __restrict__ offs) {
  if (threadIdx.x == 0) {
    int a = 0;
    for (int e = 0; e < NUM_EXPERTS; ++e) { offs[e] = a; a += cnt[e]; }
    offs[NUM_EXPERTS] = a;
  }
}

__global__ __launch_bounds__(256) void moe_fill(
    const int* __restrict__ sel, const float* __restrict__ wsel,
    const int* __restrict__ offs, int* __restrict__ cnt2,
    int* __restrict__ rows_tok, int* __restrict__ slot_of_row,
    float* __restrict__ wt_slot) {
  int t = blockIdx.x * 256 + threadIdx.x;
  if (t >= T_TOK) return;
#pragma unroll
  for (int k = 0; k < TOPK; ++k) {
    int e = sel[2 * t + k];
    int pos = atomicAdd(&cnt2[e], 1);    // list order irrelevant: per-row results
    int r = offs[e] + pos;               // depend only on the row's own token
    rows_tok[r] = t;
    slot_of_row[r] = 2 * t + k;
    wt_slot[2 * t + k] = wsel[2 * t + k];
  }
}

// ----- expert GEMM 1: inter = silu(x Wg) * (x Wu)  (bf16 out) ---------------
// Register-staged pipeline: while computing tile kt, float4 global loads for
// kt+1 are in flight. B fragments rotate: loads for n+1 issued before the
// WMMAs consuming fragment n, so dscnt waits are partial, not full drains.

__global__ __launch_bounds__(256, 1) void moe_gemm_gateup(
    const float* __restrict__ x, const float* __restrict__ w_gate,
    const float* __restrict__ w_up, const int* __restrict__ cnt,
    const int* __restrict__ offs, const int* __restrict__ rows_tok,
    unsigned short* __restrict__ inter) {
  int e = blockIdx.x / MAX_MT, mt = blockIdx.x % MAX_MT;
  int ce = cnt[e];
  if (mt * BM >= ce) return;             // block-uniform early exit
  int base = offs[e];
  int nb = blockIdx.y * BN;

  __shared__ alignas(16) unsigned short As[BM * LDT];
  __shared__ alignas(16) unsigned short Bg[BN * LDT];
  __shared__ alignas(16) unsigned short Bu[BN * LDT];
  __shared__ int toks[BM];

  int tid = threadIdx.x;
  if (tid < BM) {                        // wave-uniform predicate (waves 0,1)
    int r = mt * BM + tid;
    toks[tid] = rows_tok[base + (r < ce ? r : 0)];
  }
  __syncthreads();

  // staging geometry: A = 2 float4 per thread, B = 4 float4 per matrix
  int arow0 = tid >> 3;                  // 0..31
  int acol4 = (tid & 7) * 4;             // 0,4,...,28
  int tokA0 = toks[arow0];
  int tokA1 = toks[arow0 + 32];
  int bkl  = tid >> 5;                   // +8*j -> k row within tile
  int bil4 = (tid & 31) * 4;             // 4-col group within tile

  float4 aR[2], gR[4], uR[4];
  auto prefetchTile = [&](int kt) {
    aR[0] = *(const float4*)&x[(size_t)tokA0 * HIDDEN + kt * BK + acol4];
    aR[1] = *(const float4*)&x[(size_t)tokA1 * HIDDEN + kt * BK + acol4];
#pragma unroll
    for (int j = 0; j < 4; ++j) {
      int kl = bkl + 8 * j;
      size_t go = ((size_t)e * HIDDEN + kt * BK + kl) * INTER + nb + bil4;
      gR[j] = *(const float4*)&w_gate[go];
      uR[j] = *(const float4*)&w_up[go];
    }
  };
  auto stageTile = [&]() {
#pragma unroll
    for (int j = 0; j < 2; ++j) {
      float4 v = aR[j];
      uint2 pk; pk.x = pack2(v.x, v.y); pk.y = pack2(v.z, v.w);
      *(uint2*)&As[(arow0 + 32 * j) * LDT + acol4] = pk;
    }
#pragma unroll
    for (int j = 0; j < 4; ++j) {
      int kl = bkl + 8 * j;
      unsigned short* pg = &Bg[bil4 * LDT + kl];  // transposed store [N][K]
      unsigned short* pu = &Bu[bil4 * LDT + kl];
      float4 g = gR[j], u = uR[j];
      pg[0] = f2bf(g.x); pg[LDT] = f2bf(g.y); pg[2 * LDT] = f2bf(g.z); pg[3 * LDT] = f2bf(g.w);
      pu[0] = f2bf(u.x); pu[LDT] = f2bf(u.y); pu[2 * LDT] = f2bf(u.z); pu[3 * LDT] = f2bf(u.w);
    }
  };

  v8f cg[4] = {};
  v8f cu[4] = {};
  int lane = tid & 31, wave = tid >> 5;
  int wm = wave >> 1, wn = wave & 1;     // 4x2 wave grid over 64x128 tile

  prefetchTile(0);
  for (int kt = 0; kt < HIDDEN / BK; ++kt) {
    __syncthreads();                     // all waves done computing kt-1
    stageTile();
    __syncthreads();                     // staged tile visible
    if (kt + 1 < HIDDEN / BK) prefetchTile(kt + 1);  // overlap with WMMA

    v16bf a = load_fragA(As + wm * 16 * LDT, lane);
    // rotating fragment pipeline: loads for n+1 precede WMMAs for n
    v16bf bgCur = load_fragB(Bg + (wn * 64) * LDT, lane);
    v16bf buCur = load_fragB(Bu + (wn * 64) * LDT, lane);
#pragma unroll
    for (int n = 0; n < 4; ++n) {
      v16bf bgNow = bgCur, buNow = buCur;
      if (n < 3) {
        bgCur = load_fragB(Bg + (wn * 64 + (n + 1) * 16) * LDT, lane);
        buCur = load_fragB(Bu + (wn * 64 + (n + 1) * 16) * LDT, lane);
      }
      cg[n] = wmma_bf16(a, bgNow, cg[n]);
      cu[n] = wmma_bf16(a, buNow, cu[n]);
    }
  }

  // epilogue: silu(gate) * up -> bf16 inter rows (expert-list order)
  int half = lane >> 4, cb = lane & 15;
#pragma unroll
  for (int n = 0; n < 4; ++n) {
    int col = nb + wn * 64 + n * 16 + cb;
#pragma unroll
    for (int v = 0; v < 8; ++v) {
      int rl = wm * 16 + v + 8 * half;
      int rg = mt * BM + rl;
      if (rg < ce) {
        float g = cg[n][v], u = cu[n][v];
        float val = g / (1.f + __expf(-g)) * u;
        inter[(size_t)(base + rg) * INTER + col] = f2bf(val);
      }
    }
  }
}

// ----- expert GEMM 2: out_pair = inter Wd  (fp32 partials per pair) ---------
// A operand is already bf16 in global workspace -> staged with async
// global->LDS (double-buffered, ASYNCcnt); B register-staged like GEMM 1.

__global__ __launch_bounds__(256, 1) void moe_gemm_down(
    const unsigned short* __restrict__ inter, const float* __restrict__ w_down,
    const int* __restrict__ cnt, const int* __restrict__ offs,
    const int* __restrict__ slot_of_row, float* __restrict__ out_pair) {
  int e = blockIdx.x / MAX_MT, mt = blockIdx.x % MAX_MT;
  int ce = cnt[e];
  if (mt * BM >= ce) return;
  int base = offs[e];
  int nb = blockIdx.y * BN;
  const int KT = INTER / BK;             // 44

  __shared__ alignas(16) unsigned short As2[2][BM * LDT];
  __shared__ alignas(16) unsigned short Bs[BN * LDT];
  __shared__ int slots[BM];

  int tid = threadIdx.x;
  if (tid < BM) {
    int r = mt * BM + tid;
    slots[tid] = slot_of_row[base + (r < ce ? r : 0)];
  }

  // async A staging: one 16B chunk per thread covers the 64x32 bf16 tile
  int arow = tid >> 2;                   // 0..63
  int aseg = (tid & 3) * 8;              // bf16 elem offset within row
  int arr  = (mt * BM + arow < ce) ? (mt * BM + arow) : 0;
  const unsigned short* agbase = inter + (size_t)(base + arr) * INTER + aseg;
  auto issueA = [&](int kt) {
    async_copy_b128(agbase + kt * BK, &As2[kt & 1][arow * LDT + aseg]);
  };

  int bkl  = tid >> 5;
  int bil4 = (tid & 31) * 4;
  float4 dR[4];
  auto prefetchB = [&](int kt) {
#pragma unroll
    for (int j = 0; j < 4; ++j) {
      int kl = bkl + 8 * j;
      dR[j] = *(const float4*)&w_down[((size_t)e * INTER + kt * BK + kl) * HIDDEN + nb + bil4];
    }
  };
  auto stageB = [&]() {
#pragma unroll
    for (int j = 0; j < 4; ++j) {
      int kl = bkl + 8 * j;
      unsigned short* pb = &Bs[bil4 * LDT + kl];
      float4 d = dR[j];
      pb[0] = f2bf(d.x); pb[LDT] = f2bf(d.y); pb[2 * LDT] = f2bf(d.z); pb[3 * LDT] = f2bf(d.w);
    }
  };

  v8f c[4] = {};
  int lane = tid & 31, wave = tid >> 5;
  int wm = wave >> 1, wn = wave & 1;

  issueA(0);
  prefetchB(0);
  for (int kt = 0; kt < KT; ++kt) {
    __syncthreads();                     // compute kt-1 done; buf (kt+1)&1 free
    stageB();
    if (kt + 1 < KT) {
      issueA(kt + 1);                    // fill other buffer
      WAIT_ASYNC(1);                     // tile kt complete (kt+1 still in flight)
    } else {
      WAIT_ASYNC(0);                     // last tile complete
    }
    __syncthreads();                     // A (async) + B staged for all waves
    if (kt + 1 < KT) prefetchB(kt + 1);  // overlap next B with WMMA

    v16bf a = load_fragA(As2[kt & 1] + wm * 16 * LDT, lane);
    v16bf bCur = load_fragB(Bs + (wn * 64) * LDT, lane);
#pragma unroll
    for (int n = 0; n < 4; ++n) {
      v16bf bNow = bCur;
      if (n < 3)
        bCur = load_fragB(Bs + (wn * 64 + (n + 1) * 16) * LDT, lane);
      c[n] = wmma_bf16(a, bNow, c[n]);
    }
  }

  int half = lane >> 4, cb = lane & 15;
#pragma unroll
  for (int n = 0; n < 4; ++n) {
    int col = nb + wn * 64 + n * 16 + cb;
#pragma unroll
    for (int v = 0; v < 8; ++v) {
      int rl = wm * 16 + v + 8 * half;
      int rg = mt * BM + rl;
      if (rg < ce)
        out_pair[(size_t)slots[rl] * HIDDEN + col] = c[n][v];
    }
  }
}

// ----- final combine (deterministic, no float atomics) ----------------------

__global__ __launch_bounds__(256) void moe_combine(
    const float* __restrict__ out_pair, const float* __restrict__ wt_slot,
    float* __restrict__ out) {
  int idx = blockIdx.x * 256 + threadIdx.x;   // over T*H/4 float4s
  if (idx >= T_TOK * HIDDEN / 4) return;
  int t = idx >> 8;                           // HIDDEN/4 = 256 per token
  int h4 = idx & 255;
  float w0 = wt_slot[2 * t], w1 = wt_slot[2 * t + 1];
  float4 a = ((const float4*)(out_pair + (size_t)(2 * t) * HIDDEN))[h4];
  float4 b = ((const float4*)(out_pair + (size_t)(2 * t + 1) * HIDDEN))[h4];
  float4 r;
  r.x = w0 * a.x + w1 * b.x;
  r.y = w0 * a.y + w1 * b.y;
  r.z = w0 * a.z + w1 * b.z;
  r.w = w0 * a.w + w1 * b.w;
  ((float4*)out)[idx] = r;
}

// ----- launcher -------------------------------------------------------------

extern "C" void kernel_launch(void* const* d_in, const int* in_sizes, int n_in,
                              void* d_out, int out_size, void* d_ws, size_t ws_size,
                              hipStream_t stream) {
  const float* x      = (const float*)d_in[0];
  const float* gate_w = (const float*)d_in[1];
  const float* w_gate = (const float*)d_in[2];
  const float* w_up   = (const float*)d_in[3];
  const float* w_down = (const float*)d_in[4];
  float* out = (float*)d_out;

  char* p = (char*)d_ws;
  auto carve = [&](size_t bytes) {
    char* q = p;
    p += (bytes + 255) & ~(size_t)255;
    return (void*)q;
  };
  int*   cnt         = (int*)carve(NUM_EXPERTS * sizeof(int));
  int*   cnt2        = (int*)carve(NUM_EXPERTS * sizeof(int));
  int*   offs        = (int*)carve((NUM_EXPERTS + 1) * sizeof(int));
  int*   sel         = (int*)carve((size_t)NPAIR * sizeof(int));
  float* wsel        = (float*)carve((size_t)NPAIR * sizeof(float));
  int*   rows_tok    = (int*)carve((size_t)NPAIR * sizeof(int));
  int*   slot_of_row = (int*)carve((size_t)NPAIR * sizeof(int));
  float* wt_slot     = (float*)carve((size_t)NPAIR * sizeof(float));
  unsigned short* inter = (unsigned short*)carve((size_t)NPAIR * INTER * 2);   // 11.5 MB
  float* out_pair    = (float*)carve((size_t)NPAIR * HIDDEN * sizeof(float));  // 16.8 MB

  moe_init<<<1, 64, 0, stream>>>(cnt, cnt2);
  moe_router<<<T_TOK / 256, 256, 0, stream>>>(x, gate_w, cnt, sel, wsel);
  moe_scan<<<1, 32, 0, stream>>>(cnt, offs);
  moe_fill<<<T_TOK / 256, 256, 0, stream>>>(sel, wsel, offs, cnt2,
                                            rows_tok, slot_of_row, wt_slot);
  dim3 g1(NUM_EXPERTS * MAX_MT, INTER / BN);   // 256 x 11
  moe_gemm_gateup<<<g1, 256, 0, stream>>>(x, w_gate, w_up, cnt, offs,
                                          rows_tok, inter);
  dim3 g2(NUM_EXPERTS * MAX_MT, HIDDEN / BN);  // 256 x 8
  moe_gemm_down<<<g2, 256, 0, stream>>>(inter, w_down, cnt, offs,
                                        slot_of_row, out_pair);
  moe_combine<<<(T_TOK * HIDDEN / 4) / 256, 256, 0, stream>>>(out_pair, wt_slot, out);
}
